// TransBlock_57294863729153
// MI455X (gfx1250) — matrix operand order repaired and checked by reading, not compiled, not run on hardware
//
#include <hip/hip_runtime.h>
#include <hip/hip_bf16.h>
#include <math.h>

typedef __attribute__((ext_vector_type(16))) _Float16 v16h;
typedef __attribute__((ext_vector_type(8)))  float    v8f;

#define NN 10000
#define NE 160000
#define EPSc 1e-5f

// LDS strides (odd to dodge bank conflicts)
#define LDW  225   // w buffer (16 x 224)
#define LDA  65    // 64-wide buffers
#define LD96 97    // 96-wide buffers
#define LDDV 129   // 128-wide buffer
#define BIGSZ 2080 // shared region: act ping-pong (2x16x65) | ds (16x97) | dvd (16x129)

// ---------------- WMMA fragment helpers (CDNA5 §7.12.2 layouts) ----------------

__device__ __forceinline__ int a_kidx(int e, int lane) {
  // 16-bit A matrix 16x32: lanes 0-15 / 16-31 both hold M=0..15.
  // element e : K = hi*16 + e (contiguous, matches ldB_t)
  int hi = (lane >> 4) & 1;
  return hi * 16 + e;
}

// A fragment from a row-major f32 LDS buffer, converted to f16
__device__ __forceinline__ v16h ldA_f(const float* base, int lda, int k0, int lane) {
  int m = lane & 15;
  v16h a;
#pragma unroll
  for (int e = 0; e < 16; ++e)
    a[e] = (_Float16)base[m * lda + k0 + a_kidx(e, lane)];
  return a;
}

// B fragment from pre-transposed f16 weights wt[n][k] (K-contiguous):
// lane reads 16 contiguous halves -> single 32B vector load, no converts.
__device__ __forceinline__ v16h ldB_t(const _Float16* wt, int K, int k0, int n0, int lane) {
  int n = n0 + (lane & 15);
  int kb = (lane & 16) ? 16 : 0;
  return *(const v16h*)(wt + (size_t)n * K + k0 + kb);
}

__device__ __forceinline__ v8f vzero() {
  v8f z;
#pragma unroll
  for (int i = 0; i < 8; ++i) z[i] = 0.f;
  return z;
}

#define WMMA_F16(A, B, C) \
  __builtin_amdgcn_wmma_f32_16x16x32_f16(false, (A), false, (B), (short)0, (C), false, false)

__device__ __forceinline__ void row_stats(const float* act, float* stat, int lane) {
  if (lane < 16) {
    float s = 0.f, ss = 0.f;
    for (int k = 0; k < 64; ++k) { float x = act[lane * LDA + k]; s += x; ss += x * x; }
    float mean = s * (1.f / 64.f);
    stat[lane] = mean;
    stat[16 + lane] = rsqrtf(ss * (1.f / 64.f) - mean * mean + EPSc);
  }
}

__device__ __forceinline__ float reduce64(float* red, int t, float v) {
  red[t] = v; __syncthreads();
  if (t < 32) red[t] += red[t + 32]; __syncthreads();
  if (t < 16) red[t] += red[t + 16]; __syncthreads();
  if (t < 8)  red[t] += red[t + 8];  __syncthreads();
  if (t < 4)  red[t] += red[t + 4];  __syncthreads();
  if (t < 2)  red[t] += red[t + 2];  __syncthreads();
  if (t < 1)  red[t] += red[t + 1];  __syncthreads();
  float r = red[0]; __syncthreads();
  return r;
}

__device__ __forceinline__ float sigm(float x) { return 1.f / (1.f + __expf(-x)); }

// ---------------- Kernel 0: weight prep (f32 -> f16, transpose to K-contiguous) ----------------

__global__ void wprep_kernel(const float* __restrict__ rw1, const float* __restrict__ rw2,
                             const float* __restrict__ rw3, const float* __restrict__ lin_ws,
                             const float* __restrict__ lin_wv,
                             _Float16* rw1t, _Float16* rw2t, _Float16* rw3t,
                             _Float16* lin_wst, _Float16* lin_wvt) {
  int i = blockIdx.x * blockDim.x + threadIdx.x;
  if (i < 64 * 64) {            // 64x64, t[n*64+k] = w[k*64+n]
    int n = i >> 6, k = i & 63;
    rw1t[i] = (_Float16)rw1[k * 64 + n];
    rw2t[i] = (_Float16)rw2[k * 64 + n];
  }
  if (i < 224 * 64) {           // rw3 (64 x 224) -> t[n*64+k]
    int n = i >> 6, k = i & 63;
    rw3t[i] = (_Float16)rw3[k * 224 + n];
  }
  if (i < 128 * 96) {           // lin_ws (96 x 128) -> t[n*96+k]
    int n = i / 96, k = i - n * 96;
    lin_wst[i] = (_Float16)lin_ws[k * 128 + n];
  }
  if (i < 32 * 128) {           // lin_wv (128 x 32) -> t[n*128+k]
    int n = i >> 7, k = i & 127;
    lin_wvt[i] = (_Float16)lin_wv[k * 32 + n];
  }
}

// ---------------- Kernel 1: node prep (eq_ln1 + src/dst projections) ----------------

__launch_bounds__(256)
__global__ void node_prep_kernel(
    const float* __restrict__ node_input,
    const float* __restrict__ ln1_w0, const float* __restrict__ ln1_b0,
    const float* __restrict__ ln1_w1,
    const float* __restrict__ msrc_ws, const float* __restrict__ msrc_wv,
    const float* __restrict__ mdst_ws, const float* __restrict__ mdst_bs,
    const float* __restrict__ mdst_wv,
    float* __restrict__ src_s, float* __restrict__ src_v,
    float* __restrict__ dst_s, float* __restrict__ dst_v) {
  __shared__ float sn[4][64];
  __shared__ float vn[4][96];
  __shared__ float red[4][64];
  const int sub = threadIdx.x >> 6;
  const int t = threadIdx.x & 63;
  const int node = blockIdx.x * 4 + sub;
  const float* xi = node_input + (size_t)node * 160;

  float sval = xi[t];
  float mean = reduce64(red[sub], t, sval) * (1.f / 64.f);
  float xc = sval - mean;
  float var = reduce64(red[sub], t, xc * xc) * (1.f / 64.f);
  sn[sub][t] = xc * rsqrtf(var + EPSc) * ln1_w0[t] + ln1_b0[t];

  float v1 = xi[64 + t];
  float v2 = (t < 32) ? xi[128 + t] : 0.f;
  float vss = reduce64(red[sub], t, v1 * v1 + v2 * v2);
  float rv = rsqrtf(vss * (1.f / 96.f) + EPSc);
  vn[sub][t] = v1 * rv * ln1_w1[t / 3];
  if (t < 32) vn[sub][64 + t] = v2 * rv * ln1_w1[(64 + t) / 3];
  __syncthreads();

  float a1 = 0.f, a2 = mdst_bs[t];
  for (int k = 0; k < 64; ++k) {
    float s = sn[sub][k];
    a1 += s * msrc_ws[k * 64 + t];
    a2 += s * mdst_ws[k * 64 + t];
  }
  src_s[(size_t)node * 64 + t] = a1;
  dst_s[(size_t)node * 64 + t] = a2;
  if (t < 32) {
#pragma unroll
    for (int d = 0; d < 3; ++d) {
      float b1 = 0.f, b2 = 0.f;
      for (int u = 0; u < 32; ++u) {
        float vu = vn[sub][u * 3 + d];
        b1 += vu * msrc_wv[u * 32 + t];
        b2 += vu * mdst_wv[u * 32 + t];
      }
      src_v[(size_t)node * 96 + t * 3 + d] = b1;
      dst_v[(size_t)node * 96 + t * 3 + d] = b2;
    }
  }
}

// ---------------- Kernel 2: fused edge pipeline (WMMA heavy) ----------------
// One wave32 per 16-edge tile; 4 waves per block.

__launch_bounds__(128)
__global__ void edge_kernel(
    const float* __restrict__ edge_vec, const float* __restrict__ edge_scalars,
    const int* __restrict__ edge_src, const int* __restrict__ edge_dst,
    const _Float16* __restrict__ rw1t, const float* __restrict__ rb1,
    const float* __restrict__ rg1, const float* __restrict__ rbe1,
    const _Float16* __restrict__ rw2t, const float* __restrict__ rb2,
    const float* __restrict__ rg2, const float* __restrict__ rbe2,
    const _Float16* __restrict__ rw3t, const float* __restrict__ roff,
    const _Float16* __restrict__ lin_wst, const float* __restrict__ lin_bs,
    const _Float16* __restrict__ lin_wvt, const float* __restrict__ alpha_dot,
    const float* __restrict__ src_s, const float* __restrict__ src_v,
    const float* __restrict__ dst_s, const float* __restrict__ dst_v,
    float* __restrict__ a_logit, float* __restrict__ vs_buf,
    float* __restrict__ vv_buf) {
  __shared__ float sm_w[4][16 * LDW];
  __shared__ float sm_ms[4][16 * LDA];
  __shared__ float sm_mv[4][16 * LD96];
  __shared__ float sm_big[4][BIGSZ];
  __shared__ float sm_alpha[4][16 * LDA];
  __shared__ float sm_y[4][48];
  __shared__ float sm_stat[4][32];
  __shared__ int sm_idx[4][32];

  const int wv = threadIdx.x >> 5;
  const int lane = threadIdx.x & 31;
  const int e0 = blockIdx.x * 64 + wv * 16;
  const int mo = (lane >> 4) * 8;     // C-frag row offset
  const int ncol = lane & 15;         // C-frag column

  float* W = sm_w[wv];
  float* MS = sm_ms[wv];
  float* MV = sm_mv[wv];
  float* ACT0 = sm_big[wv];
  float* ACT1 = sm_big[wv] + 16 * LDA;
  float* DSb = sm_big[wv];
  float* DVD = sm_big[wv];
  float* ALPHA = sm_alpha[wv];
  float* Y = sm_y[wv];
  float* STAT = sm_stat[wv];
  int* IDX = sm_idx[wv];

  // prefetch next tile's edge_scalars into the cache hierarchy
  if (lane == 0 && e0 + 64 < NE)
    __builtin_prefetch(edge_scalars + (size_t)(e0 + 64) * 64, 0, 1);

  // ---- gather endpoints, Y1 = sqrt3 * rhat ----
  if (lane < 16) {
    IDX[lane] = edge_src[e0 + lane];
    IDX[16 + lane] = edge_dst[e0 + lane];
    float vx = edge_vec[(size_t)(e0 + lane) * 3 + 0];
    float vy = edge_vec[(size_t)(e0 + lane) * 3 + 1];
    float vz = edge_vec[(size_t)(e0 + lane) * 3 + 2];
    float rn = rsqrtf(vx * vx + vy * vy + vz * vz + 1e-12f);
    Y[lane * 3 + 0] = 1.7320508075688772f * vx * rn;
    Y[lane * 3 + 1] = 1.7320508075688772f * vy * rn;
    Y[lane * 3 + 2] = 1.7320508075688772f * vz * rn;
  }
  // ---- stage edge_scalars tile (16x64) into ACT1 with coalesced float4 loads ----
  {
    const float4* es4 = (const float4*)(edge_scalars + (size_t)e0 * 64);
    for (int i = lane; i < 256; i += 32) {
      int m = i >> 4, c4 = i & 15;
      float4 v = es4[m * 16 + c4];
      float* dst = &ACT1[m * LDA + c4 * 4];
      dst[0] = v.x; dst[1] = v.y; dst[2] = v.z; dst[3] = v.w;
    }
  }
  __syncthreads();
  // ---- gather ms / mv rows with float4 loads ----
  {
    const float4* ss4 = (const float4*)src_s;
    const float4* ds4 = (const float4*)dst_s;
    for (int i = lane; i < 256; i += 32) {
      int m = i >> 4, c4 = i & 15;
      float4 a = ss4[(size_t)IDX[m] * 16 + c4];
      float4 b = ds4[(size_t)IDX[16 + m] * 16 + c4];
      float* dst = &MS[m * LDA + c4 * 4];
      dst[0] = a.x + b.x; dst[1] = a.y + b.y; dst[2] = a.z + b.z; dst[3] = a.w + b.w;
    }
    const float4* sv4 = (const float4*)src_v;
    const float4* dv4 = (const float4*)dst_v;
    for (int i = lane; i < 16 * 24; i += 32) {
      int m = i / 24, c4 = i - m * 24;
      float4 a = sv4[(size_t)IDX[m] * 24 + c4];
      float4 b = dv4[(size_t)IDX[16 + m] * 24 + c4];
      float* dst = &MV[m * LD96 + c4 * 4];
      dst[0] = a.x + b.x; dst[1] = a.y + b.y; dst[2] = a.z + b.z; dst[3] = a.w + b.w;
    }
  }
  __syncthreads();

  // ---- radial MLP layer 1: ACT0 = es @ rw1 + rb1 ----
  {
    v16h A0 = ldA_f(ACT1, LDA, 0, lane);
    v16h A1 = ldA_f(ACT1, LDA, 32, lane);
    v8f acc[4];
#pragma unroll
    for (int q = 0; q < 4; ++q) {
      acc[q] = vzero();
      v16h B0 = ldB_t(rw1t, 64, 0, q * 16, lane);
      acc[q] = WMMA_F16(A0, B0, acc[q]);
      v16h B1 = ldB_t(rw1t, 64, 32, q * 16, lane);
      acc[q] = WMMA_F16(A1, B1, acc[q]);
    }
#pragma unroll
    for (int q = 0; q < 4; ++q) {
      float bias = rb1[q * 16 + ncol];
#pragma unroll
      for (int r = 0; r < 8; ++r) ACT0[(mo + r) * LDA + q * 16 + ncol] = acc[q][r] + bias;
    }
  }
  __syncthreads();
  row_stats(ACT0, STAT, lane);
  __syncthreads();
  // in-place LN + SiLU on ACT0
  for (int i = lane; i < 16 * 64; i += 32) {
    int m = i >> 6, k = i & 63;
    float y = (ACT0[m * LDA + k] - STAT[m]) * STAT[16 + m] * rg1[k] + rbe1[k];
    ACT0[m * LDA + k] = y / (1.f + __expf(-y));
  }
  __syncthreads();

  // ---- radial MLP layer 2: ACT1 = h1 @ rw2 + rb2 ----
  {
    v16h A0 = ldA_f(ACT0, LDA, 0, lane);
    v16h A1 = ldA_f(ACT0, LDA, 32, lane);
    v8f acc[4];
#pragma unroll
    for (int q = 0; q < 4; ++q) {
      acc[q] = vzero();
      v16h B0 = ldB_t(rw2t, 64, 0, q * 16, lane);
      acc[q] = WMMA_F16(A0, B0, acc[q]);
      v16h B1 = ldB_t(rw2t, 64, 32, q * 16, lane);
      acc[q] = WMMA_F16(A1, B1, acc[q]);
    }
#pragma unroll
    for (int q = 0; q < 4; ++q) {
      float bias = rb2[q * 16 + ncol];
#pragma unroll
      for (int r = 0; r < 8; ++r) ACT1[(mo + r) * LDA + q * 16 + ncol] = acc[q][r] + bias;
    }
  }
  __syncthreads();
  row_stats(ACT1, STAT, lane);
  __syncthreads();
  // in-place LN + SiLU on ACT1
  for (int i = lane; i < 16 * 64; i += 32) {
    int m = i >> 6, k = i & 63;
    float y = (ACT1[m * LDA + k] - STAT[m]) * STAT[16 + m] * rg2[k] + rbe2[k];
    ACT1[m * LDA + k] = y / (1.f + __expf(-y));
  }
  __syncthreads();

  // ---- radial MLP layer 3: W = h2 @ rw3 + roff (14 n-tiles, A held in registers) ----
  {
    v16h A0 = ldA_f(ACT1, LDA, 0, lane);
    v16h A1 = ldA_f(ACT1, LDA, 32, lane);
    for (int ch = 0; ch < 2; ++ch) {
      v8f acc[7];
#pragma unroll
      for (int q = 0; q < 7; ++q) {
        int nt = ch * 7 + q;
        acc[q] = vzero();
        v16h B0 = ldB_t(rw3t, 64, 0, nt * 16, lane);
        acc[q] = WMMA_F16(A0, B0, acc[q]);
        v16h B1 = ldB_t(rw3t, 64, 32, nt * 16, lane);
        acc[q] = WMMA_F16(A1, B1, acc[q]);
      }
#pragma unroll
      for (int q = 0; q < 7; ++q) {
        int nt = ch * 7 + q;
        float bias = roff[nt * 16 + ncol];
#pragma unroll
        for (int r = 0; r < 8; ++r) W[(mo + r) * LDW + nt * 16 + ncol] = acc[q][r] + bias;
      }
    }
  }
  __syncthreads();

  // ---- tensor product scalar path: ds = [w1*ms , w4*(mv.Y1)/sqrt3] (16 x 96) ----
  for (int i = lane; i < 16 * 96; i += 32) {
    int m = i / 96, j = i - m * 96;
    float val;
    if (j < 64) {
      val = W[m * LDW + j] * MS[m * LDA + j];
    } else {
      int k = j - 64;
      float dot = MV[m * LD96 + k * 3 + 0] * Y[m * 3 + 0] +
                  MV[m * LD96 + k * 3 + 1] * Y[m * 3 + 1] +
                  MV[m * LD96 + k * 3 + 2] * Y[m * 3 + 2];
      val = W[m * LDW + 160 + k] * dot * 0.5773502691896258f;
    }
    DSb[m * LD96 + j] = val;
  }
  __syncthreads();

  // ---- ls = ds @ lin_ws + lin_bs : tiles 0..3 -> alpha (LDS), 4..7 -> vs (global) ----
  {
    v16h A0 = ldA_f(DSb, LD96, 0, lane);
    v16h A1 = ldA_f(DSb, LD96, 32, lane);
    v16h A2 = ldA_f(DSb, LD96, 64, lane);
    for (int ch = 0; ch < 2; ++ch) {
      v8f acc[4];
#pragma unroll
      for (int q = 0; q < 4; ++q) {
        int nt = ch * 4 + q;
        acc[q] = vzero();
        v16h B0 = ldB_t(lin_wst, 96, 0, nt * 16, lane);
        acc[q] = WMMA_F16(A0, B0, acc[q]);
        v16h B1 = ldB_t(lin_wst, 96, 32, nt * 16, lane);
        acc[q] = WMMA_F16(A1, B1, acc[q]);
        v16h B2 = ldB_t(lin_wst, 96, 64, nt * 16, lane);
        acc[q] = WMMA_F16(A2, B2, acc[q]);
      }
#pragma unroll
      for (int q = 0; q < 4; ++q) {
        int nt = ch * 4 + q;
        float bias = lin_bs[nt * 16 + ncol];
        if (ch == 0) {
#pragma unroll
          for (int r = 0; r < 8; ++r) ALPHA[(mo + r) * LDA + nt * 16 + ncol] = acc[q][r] + bias;
        } else {
#pragma unroll
          for (int r = 0; r < 8; ++r)
            vs_buf[(size_t)(e0 + mo + r) * 64 + (nt - 4) * 16 + ncol] = acc[q][r] + bias;
        }
      }
    }
  }
  __syncthreads();

  // ---- vector path per dim: dv_d = [w2*ms*Y1d , w3*mv_d , w5*cross_d/sqrt2]; lv_d = dv_d @ lin_wv ----
  for (int d = 0; d < 3; ++d) {
    int d1 = (d == 2) ? 0 : d + 1;
    int d2 = (d == 0) ? 2 : d - 1;
    for (int i = lane; i < 16 * 128; i += 32) {
      int m = i >> 7, j = i & 127;
      float val;
      if (j < 64) {
        val = W[m * LDW + 64 + j] * MS[m * LDA + j] * Y[m * 3 + d];
      } else if (j < 96) {
        int k = j - 64;
        val = W[m * LDW + 128 + k] * MV[m * LD96 + k * 3 + d];
      } else {
        int k = j - 96;
        float cr = MV[m * LD96 + k * 3 + d1] * Y[m * 3 + d2] -
                   MV[m * LD96 + k * 3 + d2] * Y[m * 3 + d1];
        val = W[m * LDW + 192 + k] * cr * 0.7071067811865475f;
      }
      DVD[m * LDDV + j] = val;
    }
    __syncthreads();
    v8f acc0 = vzero(), acc1 = vzero();
#pragma unroll
    for (int k0 = 0; k0 < 128; k0 += 32) {
      v16h A = ldA_f(DVD, LDDV, k0, lane);
      v16h B0 = ldB_t(lin_wvt, 128, k0, 0, lane);
      acc0 = WMMA_F16(A, B0, acc0);
      v16h B1 = ldB_t(lin_wvt, 128, k0, 16, lane);
      acc1 = WMMA_F16(A, B1, acc1);
    }
#pragma unroll
    for (int r = 0; r < 8; ++r) {
      vv_buf[(size_t)(e0 + mo + r) * 96 + d * 32 + ncol] = acc0[r];
      vv_buf[(size_t)(e0 + mo + r) * 96 + d * 32 + 16 + ncol] = acc1[r];
    }
    __syncthreads();
  }

  // ---- attention logits: a[e,h] = sum_k smooth_leaky(alpha[e,h,k]) * alpha_dot[h,k] ----
  for (int i = lane; i < 128; i += 32) {
    int m = i >> 3, h = i & 7;
    float s = 0.f;
#pragma unroll
    for (int k = 0; k < 8; ++k) {
      float x = ALPHA[m * LDA + h * 8 + k];
      float sl = 0.6f * x + 0.4f * x * (2.f * sigm(x) - 1.f);
      s += sl * alpha_dot[h * 8 + k];
    }
    a_logit[(size_t)(e0 + m) * 8 + h] = s;
  }
}

// ---------------- Kernels 3-5: segment softmax (init, max, exp+scatter) ----------------

__global__ void init_kernel(float* amax, float* denom, float* ns, float* nv) {
  int i = blockIdx.x * blockDim.x + threadIdx.x;
  if (i < NN * 8) { amax[i] = -1e30f; denom[i] = 0.f; }
  if (i < NN * 64) ns[i] = 0.f;
  if (i < NN * 96) nv[i] = 0.f;
}

__device__ __forceinline__ void atomicMaxF(float* addr, float val) {
  unsigned int* ua = (unsigned int*)addr;
  unsigned int cur = __float_as_uint(*addr);
  while (__uint_as_float(cur) < val) {
    unsigned int prev = atomicCAS(ua, cur, __float_as_uint(val));
    if (prev == cur) break;
    cur = prev;
  }
}

__global__ void amax_kernel(const float* __restrict__ a_logit,
                            const int* __restrict__ edge_dst, float* amax) {
  int i = blockIdx.x * blockDim.x + threadIdx.x;
  if (i >= NE * 8) return;
  int e = i >> 3, h = i & 7;
  atomicMaxF(&amax[(size_t)edge_dst[e] * 8 + h], a_logit[i]);
}

__global__ void scatter_kernel(const float* __restrict__ a_logit,
                               const int* __restrict__ edge_dst,
                               const float* __restrict__ amax,
                               const float* __restrict__ vs_buf,
                               const float* __restrict__ vv_buf,
                               float* denom, float* ns, float* nv) {
  int i = blockIdx.x * blockDim.x + threadIdx.x;
  if (i >= NE * 8) return;
  int e = i >> 3, h = i & 7;
  int dn = edge_dst[e];
  float p = __expf(a_logit[i] - amax[(size_t)dn * 8 + h]);
  atomicAdd(&denom[(size_t)dn * 8 + h], p);
  const float* vsr = vs_buf + (size_t)e * 64 + h * 8;
  float* nsd = ns + (size_t)dn * 64 + h * 8;
#pragma unroll
  for (int j = 0; j < 8; ++j) atomicAdd(&nsd[j], p * vsr[j]);
#pragma unroll
  for (int dd = 0; dd < 3; ++dd) {
    const float* vvr = vv_buf + (size_t)e * 96 + dd * 32 + h * 4;
    float* nvd = nv + (size_t)dn * 96 + dd * 32 + h * 4;
#pragma unroll
    for (int j = 0; j < 4; ++j) atomicAdd(&nvd[j], p * vvr[j]);
  }
}

// ---------------- Kernel 6: node finalize (proj + residual + eq_ln2 + gated FFN) ----------------

__launch_bounds__(256)
__global__ void node_out_kernel(
    const float* __restrict__ node_input, const float* __restrict__ node_attr,
    const float* __restrict__ ln2_w0, const float* __restrict__ ln2_b0,
    const float* __restrict__ ln2_w1,
    const float* __restrict__ proj_ws, const float* __restrict__ proj_bs,
    const float* __restrict__ proj_wv,
    const float* __restrict__ f1_ws, const float* __restrict__ f1_bs,
    const float* __restrict__ f1_wv,
    const float* __restrict__ f2_ws, const float* __restrict__ f2_bs,
    const float* __restrict__ f2_wv,
    const float* __restrict__ ns_acc, const float* __restrict__ nv_acc,
    const float* __restrict__ denom, float* __restrict__ out) {
  __shared__ float nsb[4][64], nvb[4][96], xb[4][160];
  __shared__ float s2b[4][64], v2b[4][96];
  __shared__ float fsb[4][192], fvb[4][192];
  __shared__ float red[4][64];
  const int sub = threadIdx.x >> 6;
  const int t = threadIdx.x & 63;
  const int node = blockIdx.x * 4 + sub;

  nsb[sub][t] = ns_acc[(size_t)node * 64 + t] /
                (denom[(size_t)node * 8 + (t >> 3)] + 1e-12f);
  nvb[sub][t] = nv_acc[(size_t)node * 96 + t] /
                (denom[(size_t)node * 8 + ((t & 31) >> 2)] + 1e-12f);
  if (t < 32)
    nvb[sub][64 + t] = nv_acc[(size_t)node * 96 + 64 + t] /
                       (denom[(size_t)node * 8 + (t >> 2)] + 1e-12f);
  __syncthreads();

  float ps = proj_bs[t];
  for (int k = 0; k < 64; ++k) ps += nsb[sub][k] * proj_ws[k * 64 + t];
  xb[sub][t] = node_input[(size_t)node * 160 + t] + ps;
  if (t < 32) {
#pragma unroll
    for (int d = 0; d < 3; ++d) {
      float pv = 0.f;
      for (int u = 0; u < 32; ++u) pv += nvb[sub][d * 32 + u] * proj_wv[u * 32 + t];
      xb[sub][64 + t * 3 + d] = node_input[(size_t)node * 160 + 64 + t * 3 + d] + pv;
    }
  }
  __syncthreads();

  float sval = xb[sub][t];
  float mean = reduce64(red[sub], t, sval) * (1.f / 64.f);
  float xc = sval - mean;
  float var = reduce64(red[sub], t, xc * xc) * (1.f / 64.f);
  s2b[sub][t] = xc * rsqrtf(var + EPSc) * ln2_w0[t] + ln2_b0[t];
  float v1 = xb[sub][64 + t];
  float v2 = (t < 32) ? xb[sub][128 + t] : 0.f;
  float vss = reduce64(red[sub], t, v1 * v1 + v2 * v2);
  float rv = rsqrtf(vss * (1.f / 96.f) + EPSc);
  v2b[sub][t] = v1 * rv * ln2_w1[t / 3];
  if (t < 32) v2b[sub][64 + t] = v2 * rv * ln2_w1[(64 + t) / 3];
  __syncthreads();

  float a0 = node_attr[node];

#pragma unroll
  for (int rr = 0; rr < 3; ++rr) {
    int j = t + rr * 64;
    float acc = f1_bs[j];
    for (int k = 0; k < 64; ++k) acc += s2b[sub][k] * f1_ws[k * 192 + j];
    fsb[sub][j] = acc * a0;
  }
#pragma unroll
  for (int d = 0; d < 3; ++d) {
    float acc = 0.f;
    for (int u = 0; u < 32; ++u) acc += v2b[sub][u * 3 + d] * f1_wv[u * 64 + t];
    fvb[sub][t * 3 + d] = acc * a0;
  }
  __syncthreads();

  float g = sigm(fsb[sub][128 + t]);
#pragma unroll
  for (int d = 0; d < 3; ++d) fvb[sub][t * 3 + d] *= g;
  float x0 = fsb[sub][t];       fsb[sub][t] = x0 * sigm(x0);
  float x1 = fsb[sub][64 + t];  fsb[sub][64 + t] = x1 * sigm(x1);
  __syncthreads();

  float os = f2_bs[t];
  for (int k = 0; k < 128; ++k) os += fsb[sub][k] * f2_ws[k * 64 + t];
  out[(size_t)node * 160 + t] = xb[sub][t] + os * a0;
  if (t < 32) {
#pragma unroll
    for (int d = 0; d < 3; ++d) {
      float ov = 0.f;
      for (int k = 0; k < 64; ++k) ov += fvb[sub][k * 3 + d] * f2_wv[k * 32 + t];
      out[(size_t)node * 160 + 64 + t * 3 + d] = xb[sub][64 + t * 3 + d] + ov * a0;
    }
  }
}

// ---------------- launch ----------------

extern "C" void kernel_launch(void* const* d_in, const int* in_sizes, int n_in,
                              void* d_out, int out_size, void* d_ws, size_t ws_size,
                              hipStream_t stream) {
  (void)in_sizes; (void)n_in; (void)out_size; (void)ws_size;
  const float* node_input   = (const float*)d_in[0];
  const float* node_attr    = (const float*)d_in[1];
  const float* edge_vec     = (const float*)d_in[2];
  const float* edge_scalars = (const float*)d_in[3];
  const int*   edge_src     = (const int*)d_in[4];
  const int*   edge_dst     = (const int*)d_in[5];
  const float* ln1_w0 = (const float*)d_in[6];
  const float* ln1_b0 = (const float*)d_in[7];
  const float* ln1_w1 = (const float*)d_in[8];
  const float* ln2_w0 = (const float*)d_in[9];
  const float* ln2_b0 = (const float*)d_in[10];
  const float* ln2_w1 = (const float*)d_in[11];
  const float* msrc_ws = (const float*)d_in[12];
  const float* msrc_wv = (const float*)d_in[13];
  const float* mdst_ws = (const float*)d_in[14];
  const float* mdst_bs = (const float*)d_in[15];
  const float* mdst_wv = (const float*)d_in[16];
  const float* rw1  = (const float*)d_in[17];
  const float* rb1  = (const float*)d_in[18];
  const float* rg1  = (const float*)d_in[19];
  const float* rbe1 = (const float*)d_in[20];
  const float* rw2  = (const float*)d_in[21];
  const float* rb2  = (const float*)d_in[22];
  const float* rg2  = (const float*)d_in[23];
  const float* rbe2 = (const float*)d_in[24];
  const float* rw3  = (const float*)d_in[25];
  const float* roff = (const float*)d_in[26];
  const float* lin_ws = (const float*)d_in[27];
  const float* lin_bs = (const float*)d_in[28];
  const float* lin_wv = (const float*)d_in[29];
  const float* alpha_dot = (const float*)d_in[30];
  const float* proj_ws = (const float*)d_in[31];
  const float* proj_bs = (const float*)d_in[32];
  const float* proj_wv = (const float*)d_in[33];
  const float* f1_ws = (const float*)d_in[34];
  const float* f1_bs = (const float*)d_in[35];
  const float* f1_wv = (const float*)d_in[36];
  const float* f2_ws = (const float*)d_in[37];
  const float* f2_bs = (const float*)d_in[38];
  const float* f2_wv = (const float*)d_in[39];

  float* ws = (float*)d_ws;
  size_t o = 0;
  float* src_s  = ws + o; o += (size_t)NN * 64;
  float* src_v  = ws + o; o += (size_t)NN * 96;
  float* dst_s  = ws + o; o += (size_t)NN * 64;
  float* dst_v  = ws + o; o += (size_t)NN * 96;
  float* amax   = ws + o; o += (size_t)NN * 8;
  float* denom  = ws + o; o += (size_t)NN * 8;
  float* ns_acc = ws + o; o += (size_t)NN * 64;
  float* nv_acc = ws + o; o += (size_t)NN * 96;
  float* a_logit = ws + o; o += (size_t)NE * 8;
  float* vs_buf  = ws + o; o += (size_t)NE * 64;
  float* vv_buf  = ws + o; o += (size_t)NE * 96;

  // f16 transposed weights (32B-aligned: all prior sizes are multiples of 8 floats)
  _Float16* h = (_Float16*)(ws + o);
  size_t ho = 0;
  _Float16* rw1t    = h + ho; ho += 64 * 64;
  _Float16* rw2t    = h + ho; ho += 64 * 64;
  _Float16* rw3t    = h + ho; ho += 224 * 64;
  _Float16* lin_wst = h + ho; ho += 128 * 96;
  _Float16* lin_wvt = h + ho; ho += 32 * 128;

  wprep_kernel<<<(224 * 64 + 255) / 256, 256, 0, stream>>>(
      rw1, rw2, rw3, lin_ws, lin_wv, rw1t, rw2t, rw3t, lin_wst, lin_wvt);
  init_kernel<<<(NN * 96 + 255) / 256, 256, 0, stream>>>(amax, denom, ns_acc, nv_acc);
  node_prep_kernel<<<NN / 4, 256, 0, stream>>>(
      node_input, ln1_w0, ln1_b0, ln1_w1, msrc_ws, msrc_wv, mdst_ws, mdst_bs,
      mdst_wv, src_s, src_v, dst_s, dst_v);
  edge_kernel<<<NE / 64, 128, 0, stream>>>(
      edge_vec, edge_scalars, edge_src, edge_dst, rw1t, rb1, rg1, rbe1, rw2t, rb2,
      rg2, rbe2, rw3t, roff, lin_wst, lin_bs, lin_wvt, alpha_dot, src_s, src_v,
      dst_s, dst_v, a_logit, vs_buf, vv_buf);
  amax_kernel<<<(NE * 8 + 255) / 256, 256, 0, stream>>>(a_logit, edge_dst, amax);
  scatter_kernel<<<(NE * 8 + 255) / 256, 256, 0, stream>>>(
      a_logit, edge_dst, amax, vs_buf, vv_buf, denom, ns_acc, nv_acc);
  node_out_kernel<<<NN / 4, 256, 0, stream>>>(
      node_input, node_attr, ln2_w0, ln2_b0, ln2_w1, proj_ws, proj_bs, proj_wv,
      f1_ws, f1_bs, f1_wv, f2_ws, f2_bs, f2_wv, ns_acc, nv_acc, denom,
      (float*)d_out);
}
